// ThreeGraphConvolution_65755949302355
// MI455X (gfx1250) — compile-verified, matrix operand order checked by reading, not run on hardware
//
#include <hip/hip_runtime.h>
#include <hip/hip_bf16.h>
#include <stdint.h>

// ---------------------------------------------------------------------------
// 3-layer GCN + FC for MI455X (gfx1250).
// GEMMs: v_wmma_f32_16x16x32_bf16, 16x(16*NT) tile per wave (A-fragment reuse).
// Edge aggregation: float4 gathers + native f32 global atomics.
// ---------------------------------------------------------------------------

typedef __attribute__((ext_vector_type(16))) __bf16 v16bf;
typedef __attribute__((ext_vector_type(8)))  __bf16 v8bf;
typedef __attribute__((ext_vector_type(8)))  float  v8f;

#define N_NODES 50000
#define N_EDGES 400000

// ---------------- elementwise conversion kernels ----------------

__global__ void k_f32_to_bf16(const float* __restrict__ in,
                              __bf16* __restrict__ out, long n) {
    long i = (long)blockIdx.x * blockDim.x + threadIdx.x;
    if (i < n) out[i] = (__bf16)in[i];
}

__global__ void k_relu_to_bf16(const float* __restrict__ in,
                               __bf16* __restrict__ out, long n) {
    long i = (long)blockIdx.x * blockDim.x + threadIdx.x;
    if (i < n) out[i] = (__bf16)fmaxf(in[i], 0.0f);
}

// W [Kin x Kout] row-major f32  ->  Wt [KoutPad x Kin] row-major bf16 (zero pad)
__global__ void k_transpose_to_bf16(const float* __restrict__ W,
                                    __bf16* __restrict__ Wt,
                                    int Kin, int Kout, int KoutPad) {
    long i = (long)blockIdx.x * blockDim.x + threadIdx.x;
    long total = (long)KoutPad * Kin;
    if (i >= total) return;
    int n = (int)(i / Kin);
    int k = (int)(i % Kin);
    float v = (n < Kout) ? W[(long)k * Kout + n] : 0.0f;
    Wt[i] = (__bf16)v;
}

// ---------------- degree kernels ----------------

__global__ void k_init_deg(float* __restrict__ deg, int n) {
    int i = blockIdx.x * blockDim.x + threadIdx.x;
    if (i < n) deg[i] = 1.0f;  // self loop
}

__global__ void k_count_deg(const long long* __restrict__ ei,
                            float* __restrict__ deg, int E) {
    int e = blockIdx.x * blockDim.x + threadIdx.x;
    if (e < E) {
        long long d = ei[(long)E + e];  // dst row
        unsafeAtomicAdd(&deg[(int)d], 1.0f);
    }
}

// deg -> inv_sqrt (in place) and inv_deg
__global__ void k_finalize_deg(float* __restrict__ deg_to_invsqrt,
                               float* __restrict__ inv_deg, int n) {
    int i = blockIdx.x * blockDim.x + threadIdx.x;
    if (i < n) {
        float d = deg_to_invsqrt[i];
        inv_deg[i] = 1.0f / d;
        deg_to_invsqrt[i] = rsqrtf(d);
    }
}

// ---------------- WMMA GEMM: C[N x Nout] = A[N x Kin] * Bt[Nout x Kin]^T ----
// Each wave computes a 16 x (16*NT) strip: one A fragment feeds NT WMMAs.
// Fragment layout per CDNA5 ISA (16-bit A 16x32): lane L holds row L%16,
// K halves at offsets 8*(L/16) and 16+8*(L/16). B mirrors with column L%16.

template <int NT>
__global__ void k_gemm_bf16_wmma(const __bf16* __restrict__ A,
                                 const __bf16* __restrict__ Bt,
                                 float* __restrict__ C,
                                 int N, int Kin, int Nout) {
    const int wave = threadIdx.x >> 5;
    const int lane = threadIdx.x & 31;
    const int mt = blockIdx.x * (blockDim.x >> 5) + wave;   // wave-uniform
    if (mt * 16 >= N) return;                               // whole wave exits

    const int m0   = mt * 16;
    const int n0   = blockIdx.y * (16 * NT);
    const int half = lane >> 4;   // 0|1
    const int r    = lane & 15;

    const __bf16* arow = A  + (long)(m0 + r) * Kin + half * 8;
    const __bf16* brow = Bt + (long)(n0 + r) * Kin + half * 8;
    const long    bstep = (long)16 * Kin;   // next 16-column tile of Bt

    v8f acc[NT];
    #pragma unroll
    for (int t = 0; t < NT; ++t) acc[t] = (v8f){};

    union Frag { v16bf v; v8bf h[2]; };

    for (int k = 0; k < Kin; k += 32) {
        Frag a;
        a.h[0] = *(const v8bf*)(arow + k);
        a.h[1] = *(const v8bf*)(arow + k + 16);
        Frag b[NT];
        #pragma unroll
        for (int t = 0; t < NT; ++t) {
            b[t].h[0] = *(const v8bf*)(brow + (long)t * bstep + k);
            b[t].h[1] = *(const v8bf*)(brow + (long)t * bstep + k + 16);
        }
        #pragma unroll
        for (int t = 0; t < NT; ++t) {
            acc[t] = __builtin_amdgcn_wmma_f32_16x16x32_bf16(
                /*neg_a=*/false, a.v, /*neg_b=*/false, b[t].v,
                /*c_mod=*/(short)0, acc[t], /*reuse_a=*/false, /*reuse_b=*/false);
        }
    }

    // C/D layout: VGPR v, lane L -> M = v + 8*(L/16), N = L%16
    #pragma unroll
    for (int t = 0; t < NT; ++t) {
        float* cbase = C + (long)(m0 + half * 8) * Nout + n0 + t * 16 + r;
        #pragma unroll
        for (int v = 0; v < 8; ++v)
            cbase[(long)v * Nout] = acc[t][v];
    }
}

// ---------------- aggregation kernels ----------------

// agg[i,f] = H[i,f] * inv_deg[i] + bias[f]
__global__ void k_self_init(const float* __restrict__ H,
                            const float* __restrict__ inv_deg,
                            const float* __restrict__ bias,
                            float* __restrict__ agg, int N, int F) {
    long i = (long)blockIdx.x * blockDim.x + threadIdx.x;
    long total = (long)N * F;
    if (i >= total) return;
    int node = (int)(i / F);
    int f    = (int)(i % F);
    agg[i] = H[i] * inv_deg[node] + bias[f];
}

// one wave per edge: agg[dst,:] += H[src,:] * (inv_sqrt[src]*inv_sqrt[dst])
// float4 row gathers (global_load_b128) + native f32 atomics.
__global__ void k_edge_agg(const float* __restrict__ H,
                           const long long* __restrict__ ei,
                           const float* __restrict__ inv_sqrt,
                           float* __restrict__ agg, int E, int F) {
    int e = blockIdx.x * (blockDim.x >> 5) + (threadIdx.x >> 5);
    if (e >= E) return;
    int lane = threadIdx.x & 31;
    long long s = ei[e];
    long long d = ei[(long)E + e];
    float nrm = inv_sqrt[(int)s] * inv_sqrt[(int)d];
    const float* hs = H   + (long)s * F;
    float*       ad = agg + (long)d * F;
    for (int f = lane * 4; f < F; f += 128) {   // F % 128 == 0 for all layers
        float4 h = *(const float4*)(hs + f);
        unsafeAtomicAdd(&ad[f + 0], h.x * nrm);
        unsafeAtomicAdd(&ad[f + 1], h.y * nrm);
        unsafeAtomicAdd(&ad[f + 2], h.z * nrm);
        unsafeAtomicAdd(&ad[f + 3], h.w * nrm);
    }
}

// out[i,c] = buf[i,Cpad..] + bias[c]  (final FC epilogue, drops padding)
__global__ void k_fc_epilogue(const float* __restrict__ buf,
                              const float* __restrict__ bias,
                              float* __restrict__ out,
                              int N, int Cout, int Cpad) {
    long i = (long)blockIdx.x * blockDim.x + threadIdx.x;
    long total = (long)N * Cout;
    if (i >= total) return;
    int node = (int)(i / Cout);
    int c    = (int)(i % Cout);
    out[i] = buf[(long)node * Cpad + c] + bias[c];
}

// ---------------------------------------------------------------------------

static inline long cdiv(long a, long b) { return (a + b - 1) / b; }

extern "C" void kernel_launch(void* const* d_in, const int* in_sizes, int n_in,
                              void* d_out, int out_size, void* d_ws, size_t ws_size,
                              hipStream_t stream) {
    const float*      x   = (const float*)d_in[0];
    const long long*  ei  = (const long long*)d_in[1];   // int64 [2, E]
    const float*      W1  = (const float*)d_in[2];
    const float*      b1  = (const float*)d_in[3];
    const float*      W2  = (const float*)d_in[4];
    const float*      b2  = (const float*)d_in[5];
    const float*      W3  = (const float*)d_in[6];
    const float*      b3  = (const float*)d_in[7];
    const float*      Wfc = (const float*)d_in[8];
    const float*      bfc = (const float*)d_in[9];
    float*            out = (float*)d_out;

    const int N = N_NODES;
    const int E = (n_in > 1 && in_sizes[1] > 0) ? in_sizes[1] / 2 : N_EDGES;

    // ---- workspace carve-up (aligned 256B) ----
    char* p = (char*)d_ws;
    auto carve = [&](size_t bytes) -> char* {
        char* r = p;
        p += (bytes + 255) & ~(size_t)255;
        return r;
    };
    __bf16* bufIn  = (__bf16*)carve((size_t)N * 1024 * sizeof(__bf16)); // 100 MB
    float*  bufH   = (float*) carve((size_t)N * 1024 * sizeof(float));  // 205 MB
    float*  bufAgg = (float*) carve((size_t)N * 1024 * sizeof(float));  // 205 MB
    __bf16* W1t    = (__bf16*)carve((size_t)1024 * 128 * sizeof(__bf16));
    __bf16* W2t    = (__bf16*)carve((size_t)512 * 1024 * sizeof(__bf16));
    __bf16* W3t    = (__bf16*)carve((size_t)128 * 512 * sizeof(__bf16));
    __bf16* Wfct   = (__bf16*)carve((size_t)48 * 128 * sizeof(__bf16));
    float*  invsq  = (float*) carve((size_t)N * sizeof(float));  // deg -> inv_sqrt
    float*  invdeg = (float*) carve((size_t)N * sizeof(float));
    float*  fcbuf  = (float*) carve((size_t)N * 48 * sizeof(float));

    const int TB = 256;

    // ---- preprocessing ----
    k_f32_to_bf16<<<cdiv((long)N * 128, TB), TB, 0, stream>>>(x, bufIn, (long)N * 128);
    k_transpose_to_bf16<<<cdiv((long)1024 * 128, TB), TB, 0, stream>>>(W1, W1t, 128, 1024, 1024);
    k_transpose_to_bf16<<<cdiv((long)512 * 1024, TB), TB, 0, stream>>>(W2, W2t, 1024, 512, 512);
    k_transpose_to_bf16<<<cdiv((long)128 * 512, TB), TB, 0, stream>>>(W3, W3t, 512, 128, 128);
    k_transpose_to_bf16<<<cdiv((long)48 * 128, TB), TB, 0, stream>>>(Wfc, Wfct, 128, 40, 48);

    k_init_deg<<<cdiv(N, TB), TB, 0, stream>>>(invsq, N);
    k_count_deg<<<cdiv(E, TB), TB, 0, stream>>>(ei, invsq, E);
    k_finalize_deg<<<cdiv(N, TB), TB, 0, stream>>>(invsq, invdeg, N);

    const int MT = N / 16;                     // 3125 tiles along N (N % 16 == 0)
    const int gx = (int)cdiv(MT, TB / 32);     // 8 waves (m-tiles) per block

    // ---- layer 1: 128 -> 1024 ----
    {
        const int Kin = 128, F = 1024;
        k_gemm_bf16_wmma<4><<<dim3(gx, F / 64), TB, 0, stream>>>(bufIn, W1t, bufH, N, Kin, F);
        k_self_init<<<cdiv((long)N * F, TB), TB, 0, stream>>>(bufH, invdeg, b1, bufAgg, N, F);
        k_edge_agg<<<cdiv(E, TB / 32), TB, 0, stream>>>(bufH, ei, invsq, bufAgg, E, F);
        k_relu_to_bf16<<<cdiv((long)N * F, TB), TB, 0, stream>>>(bufAgg, bufIn, (long)N * F);
    }
    // ---- layer 2: 1024 -> 512 ----
    {
        const int Kin = 1024, F = 512;
        k_gemm_bf16_wmma<4><<<dim3(gx, F / 64), TB, 0, stream>>>(bufIn, W2t, bufH, N, Kin, F);
        k_self_init<<<cdiv((long)N * F, TB), TB, 0, stream>>>(bufH, invdeg, b2, bufAgg, N, F);
        k_edge_agg<<<cdiv(E, TB / 32), TB, 0, stream>>>(bufH, ei, invsq, bufAgg, E, F);
        k_relu_to_bf16<<<cdiv((long)N * F, TB), TB, 0, stream>>>(bufAgg, bufIn, (long)N * F);
    }
    // ---- layer 3: 512 -> 128 ----
    {
        const int Kin = 512, F = 128;
        k_gemm_bf16_wmma<4><<<dim3(gx, F / 64), TB, 0, stream>>>(bufIn, W3t, bufH, N, Kin, F);
        k_self_init<<<cdiv((long)N * F, TB), TB, 0, stream>>>(bufH, invdeg, b3, bufAgg, N, F);
        k_edge_agg<<<cdiv(E, TB / 32), TB, 0, stream>>>(bufH, ei, invsq, bufAgg, E, F);
        k_relu_to_bf16<<<cdiv((long)N * F, TB), TB, 0, stream>>>(bufAgg, bufIn, (long)N * F);
    }
    // ---- FC head: 128 -> 40 (padded to 48, NT=3 -> one grid column) ----
    k_gemm_bf16_wmma<3><<<dim3(gx, 1), TB, 0, stream>>>(bufIn, Wfct, fcbuf, N, 128, 48);
    k_fc_epilogue<<<cdiv((long)N * 40, TB), TB, 0, stream>>>(fcbuf, bfc, out, N, 40, 48);
}
